// TopKAutoEncoder_18348100288732
// MI455X (gfx1250) — matrix operand order repaired and checked by reading, not compiled, not run on hardware
//
#include <hip/hip_runtime.h>
#include <hip/hip_bf16.h>

typedef __bf16 v16bf __attribute__((ext_vector_type(16)));
typedef float  v8f   __attribute__((ext_vector_type(8)));

typedef unsigned int   uint_t;
typedef unsigned short ushort_t;

// exact pointer types expected by __builtin_amdgcn_global_load_async_to_lds_b128
typedef int v4i __attribute__((vector_size(16)));
typedef __attribute__((address_space(1))) v4i* gv4i_p;   // global (AS1)
typedef __attribute__((address_space(3))) v4i* lv4i_p;   // LDS    (AS3)

#define BDIM 16384
#define DDIM 4096
#define KTOP 64

#if defined(__AMDGCN__) && \
    __has_builtin(__builtin_amdgcn_global_load_async_to_lds_b128) && \
    __has_builtin(__builtin_amdgcn_s_wait_asynccnt)
#define USE_ASYNC 1
#else
#define USE_ASYNC 0
#endif

// ---------------------------------------------------------------------------
// helpers
// ---------------------------------------------------------------------------
__device__ __forceinline__ ushort_t f2bf(float f) {
    union { float f; uint_t u; } c; c.f = f;
    uint_t u = c.u;
    u += 0x7FFFu + ((u >> 16) & 1u);   // round-to-nearest-even
    return (ushort_t)(u >> 16);
}
__device__ __forceinline__ uint_t pk2bf(float a, float b) {
    return (uint_t)f2bf(a) | ((uint_t)f2bf(b) << 16);
}

union BF16Frag { v16bf v; uint4 q[2]; };

// ---------------------------------------------------------------------------
// Pre-pass 1: Xbf[r][k] = bf16(x[r][k] - pre_bias[k])   (pre-centered A)
// ---------------------------------------------------------------------------
__global__ __launch_bounds__(256)
void prep_x_kernel(const float* __restrict__ x,
                   const float* __restrict__ pre_bias,
                   ushort_t* __restrict__ xbf)
{
    const size_t base = ((size_t)blockIdx.x * 256 + threadIdx.x) * 8;
    const int k = (int)(base & (DDIM - 1));
    const float4 v0 = *(const float4*)(x + base);
    const float4 v1 = *(const float4*)(x + base + 4);
    const float4 p0 = *(const float4*)(pre_bias + k);
    const float4 p1 = *(const float4*)(pre_bias + k + 4);
    uint4 o;
    o.x = pk2bf(v0.x - p0.x, v0.y - p0.y);
    o.y = pk2bf(v0.z - p0.z, v0.w - p0.w);
    o.z = pk2bf(v1.x - p1.x, v1.y - p1.y);
    o.w = pk2bf(v1.z - p1.z, v1.w - p1.w);
    *(uint4*)(xbf + base) = o;
}

// ---------------------------------------------------------------------------
// Pre-pass 2: Wt[n][k] = bf16(W_enc[k][n])   (K-contiguous B operand)
// ---------------------------------------------------------------------------
__global__ __launch_bounds__(256)
void prep_w_kernel(const float* __restrict__ W, ushort_t* __restrict__ wt)
{
    __shared__ ushort_t tile[32][33];
    const int tx = threadIdx.x & 31;
    const int ty = threadIdx.x >> 5;           // 0..7
    const int k0 = blockIdx.y * 32;
    const int n0 = blockIdx.x * 32;
    #pragma unroll
    for (int i = 0; i < 4; ++i) {
        const int kk = ty + i * 8;
        tile[kk][tx] = f2bf(W[(size_t)(k0 + kk) * DDIM + n0 + tx]);
    }
    __syncthreads();
    #pragma unroll
    for (int i = 0; i < 4; ++i) {
        const int nn = ty + i * 8;
        wt[(size_t)(n0 + nn) * DDIM + k0 + tx] = tile[tx][nn];
    }
}

// ---------------------------------------------------------------------------
// Kernel: acts[B,N] = Xbf @ Wt^T + latent_bias
// block tile 128(M) x 128(N), K-step 32, 256 threads = 8 waves.
// Each wave: 32(M) x 64(N) via 2x4 v_wmma_f32_16x16x32_bf16 accumulators.
// Double-buffered LDS; async global->LDS copies when available.
// ---------------------------------------------------------------------------
__global__ __launch_bounds__(256)
void enc_gemm_kernel(const ushort_t* __restrict__ xbf,   // [B][K] bf16 (centered)
                     const ushort_t* __restrict__ wt,    // [N][K] bf16
                     const float* __restrict__ lat_bias, // [N]
                     float* __restrict__ acts)           // [B][N]
{
    __shared__ ushort_t As[2][128][40];   // [buf][m][k], pad 32->40
    __shared__ ushort_t Bs[2][128][40];   // [buf][n][k], pad 32->40

    const int tid  = threadIdx.x;
    const int wave = tid >> 5;
    const int lane = tid & 31;
    const int half = lane >> 4;
    const int l16  = lane & 15;

    const int bm = blockIdx.y;
    const int bn = blockIdx.x;
    const int wm = wave & 3;              // M sub-block: 32 rows
    const int wn = wave >> 2;             // N sub-block: 64 cols
    const int row0 = bm * 128;
    const int col0 = bn * 128;

    v8f acc[2][4];
    #pragma unroll
    for (int i = 0; i < 2; ++i)
        #pragma unroll
        for (int j = 0; j < 4; ++j)
            acc[i][j] = (v8f){0.f,0.f,0.f,0.f,0.f,0.f,0.f,0.f};

    // per-thread staging coords: 512 uint4 per tile, 2 per thread
    // fi = tid + i*256 ; r = fi>>2 (row 0..127) ; c4 = fi&3 (16B chunk in 64B row)

#if USE_ASYNC
    // ---- prologue: async stage tile 0 into buffer 0 -----------------------
    #pragma unroll
    for (int i = 0; i < 2; ++i) {
        const int fi = tid + i * 256, r = fi >> 2, c4 = fi & 3;
        __builtin_amdgcn_global_load_async_to_lds_b128(
            (gv4i_p)(xbf + (size_t)(row0 + r) * DDIM + c4 * 8),
            (lv4i_p)&As[0][r][c4 * 8], 0, 0);
        __builtin_amdgcn_global_load_async_to_lds_b128(
            (gv4i_p)(wt + (size_t)(col0 + r) * DDIM + c4 * 8),
            (lv4i_p)&Bs[0][r][c4 * 8], 0, 0);
    }
    __builtin_amdgcn_s_wait_asynccnt(0);
    __syncthreads();
#else
    #pragma unroll
    for (int i = 0; i < 2; ++i) {
        const int fi = tid + i * 256, r = fi >> 2, c4 = fi & 3;
        *(uint4*)&As[0][r][c4 * 8] = *(const uint4*)(xbf + (size_t)(row0 + r) * DDIM + c4 * 8);
        *(uint4*)&Bs[0][r][c4 * 8] = *(const uint4*)(wt + (size_t)(col0 + r) * DDIM + c4 * 8);
    }
    __syncthreads();
#endif

    for (int kt = 0; kt < DDIM; kt += 32) {
        const int cur = (kt >> 5) & 1;
        const int nxt = cur ^ 1;
        const bool more = (kt + 32) < DDIM;

#if USE_ASYNC
        // fire async copies for the next K-tile while we compute this one
        if (more) {
            #pragma unroll
            for (int i = 0; i < 2; ++i) {
                const int fi = tid + i * 256, r = fi >> 2, c4 = fi & 3;
                __builtin_amdgcn_global_load_async_to_lds_b128(
                    (gv4i_p)(xbf + (size_t)(row0 + r) * DDIM + kt + 32 + c4 * 8),
                    (lv4i_p)&As[nxt][r][c4 * 8], 0, 0);
                __builtin_amdgcn_global_load_async_to_lds_b128(
                    (gv4i_p)(wt + (size_t)(col0 + r) * DDIM + kt + 32 + c4 * 8),
                    (lv4i_p)&Bs[nxt][r][c4 * 8], 0, 0);
            }
        }
#else
        uint4 ra[2], rb[2];
        if (more) {
            #pragma unroll
            for (int i = 0; i < 2; ++i) {
                const int fi = tid + i * 256, r = fi >> 2, c4 = fi & 3;
                ra[i] = *(const uint4*)(xbf + (size_t)(row0 + r) * DDIM + kt + 32 + c4 * 8);
                rb[i] = *(const uint4*)(wt  + (size_t)(col0 + r) * DDIM + kt + 32 + c4 * 8);
            }
        }
#endif

        // ---- compute on current buffer ------------------------------------
        BF16Frag afr[2], bfr[4];
        #pragma unroll
        for (int s = 0; s < 2; ++s) {
            const int r = wm * 32 + s * 16 + l16;
            // lanes 0-15: K 0..7 then 16..23 ; lanes 16-31: K 8..15 then 24..31
            afr[s].q[0] = *(const uint4*)&As[cur][r][half * 8];
            afr[s].q[1] = *(const uint4*)&As[cur][r][16 + half * 8];
        }
        #pragma unroll
        for (int t = 0; t < 4; ++t) {
            const int n = wn * 64 + t * 16 + l16;
            // lanes 0-15: K 0..7 then 16..23 ; lanes 16-31: K 8..15 then 24..31
            bfr[t].q[0] = *(const uint4*)&Bs[cur][n][half * 8];
            bfr[t].q[1] = *(const uint4*)&Bs[cur][n][16 + half * 8];
        }
        #pragma unroll
        for (int i = 0; i < 2; ++i)
            #pragma unroll
            for (int j = 0; j < 4; ++j)
                acc[i][j] = __builtin_amdgcn_wmma_f32_16x16x32_bf16(
                    false, afr[i].v, false, bfr[j].v,
                    (short)0, acc[i][j], false, false);

#if USE_ASYNC
        __builtin_amdgcn_s_wait_asynccnt(0);
        __syncthreads();
#else
        if (more) {
            #pragma unroll
            for (int i = 0; i < 2; ++i) {
                const int fi = tid + i * 256, r = fi >> 2, c4 = fi & 3;
                *(uint4*)&As[nxt][r][c4 * 8] = ra[i];
                *(uint4*)&Bs[nxt][r][c4 * 8] = rb[i];
            }
        }
        __syncthreads();
#endif
    }

    // ---- epilogue: D-matrix layout -> global, add latent_bias -------------
    #pragma unroll
    for (int i = 0; i < 2; ++i) {
        #pragma unroll
        for (int j = 0; j < 4; ++j) {
            const int cbase = col0 + wn * 64 + j * 16 + l16;
            const float lb  = lat_bias[cbase];
            #pragma unroll
            for (int r = 0; r < 8; ++r) {
                const int rr = row0 + wm * 32 + i * 16 + r + 8 * half;
                acts[(size_t)rr * DDIM + cbase] = acc[i][j][r] + lb;
            }
        }
    }
}

// ---------------------------------------------------------------------------
// Kernel: per-row top-64 via MSB-first radix select on order-preserving
// uint keys, then compact (idx, relu(val)).  1 block (256 thr) per row.
// ---------------------------------------------------------------------------
__global__ __launch_bounds__(256)
void topk_kernel(const float* __restrict__ acts,
                 float* __restrict__ vals,   // [B, 64]
                 int*   __restrict__ idxs)   // [B, 64]
{
    __shared__ uint_t keys[DDIM];
    __shared__ uint_t hist[256];
    __shared__ uint_t s_prefix, s_want, s_cnt, s_tie;

    const int    tid = threadIdx.x;
    const size_t row = blockIdx.x;
    const float* arow = acts + row * DDIM;

    for (int i = tid; i < DDIM; i += 256) {
        union { float f; uint_t u; } c; c.f = arow[i];
        const uint_t u = c.u;
        keys[i] = (u & 0x80000000u) ? ~u : (u | 0x80000000u);
    }
    if (tid == 0) { s_prefix = 0u; s_want = KTOP; }
    __syncthreads();

    #pragma unroll
    for (int shift = 24; shift >= 0; shift -= 8) {
        hist[tid] = 0u;
        __syncthreads();
        const uint_t pfx = s_prefix;
        const uint_t hm  = (shift == 24) ? 0u : (0xFFFFFFFFu << (shift + 8));
        for (int i = tid; i < DDIM; i += 256) {
            const uint_t key = keys[i];
            if ((key & hm) == (pfx & hm))
                atomicAdd(&hist[(key >> shift) & 0xFFu], 1u);
        }
        __syncthreads();
        if (tid == 0) {
            uint_t want = s_want, cum = 0u;
            int b = 255;
            for (; b > 0; --b) {
                const uint_t c = hist[b];
                if (cum + c >= want) break;
                cum += c;
            }
            s_want   = want - cum;
            s_prefix = pfx | ((uint_t)b << shift);
        }
        __syncthreads();
    }

    const uint_t T        = s_prefix;
    const uint_t wantTies = s_want;
    if (tid == 0) { s_cnt = 0u; s_tie = 0u; }
    __syncthreads();

    for (int i = tid; i < DDIM; i += 256) {
        const uint_t key = keys[i];
        bool take = false;
        if (key > T) take = true;
        else if (key == T) {
            const uint_t t = atomicAdd(&s_tie, 1u);
            take = (t < wantTies);
        }
        if (take) {
            const uint_t p = atomicAdd(&s_cnt, 1u);
            if (p < (uint_t)KTOP) {
                const uint_t u = (key & 0x80000000u) ? (key ^ 0x80000000u) : ~key;
                union { uint_t u; float f; } c; c.u = u;
                const float f = c.f;
                vals[row * KTOP + p] = f > 0.f ? f : 0.f;   // ReLU after top-k
                idxs[row * KTOP + p] = i;
            }
        }
    }
}

// ---------------------------------------------------------------------------
// Kernel: out[row,:] = sum_j val_j * W_dec[idx_j,:] + pre_bias
// 1 block (256 thr) per row; W_dec (64 MB) stays L2-resident.
// ---------------------------------------------------------------------------
__global__ __launch_bounds__(256)
void dec_gemm_kernel(const float* __restrict__ vals,
                     const int*   __restrict__ idxs,
                     const float* __restrict__ W_dec,     // [D, D] row-major
                     const float* __restrict__ pre_bias,  // [D]
                     float* __restrict__ out)             // [B, D]
{
    __shared__ float sv[KTOP];
    __shared__ int   si[KTOP];

    const int    tid = threadIdx.x;
    const size_t row = blockIdx.x;

    if (tid < KTOP) {
        sv[tid] = vals[row * KTOP + tid];
        si[tid] = idxs[row * KTOP + tid];
    }
    __syncthreads();

    float4 acc[4];
    #pragma unroll
    for (int i = 0; i < 4; ++i) acc[i] = make_float4(0.f, 0.f, 0.f, 0.f);

    for (int j = 0; j < KTOP; ++j) {
        const float v = sv[j];
        const float4* wr = (const float4*)(W_dec + (size_t)si[j] * DDIM);
        #pragma unroll
        for (int i = 0; i < 4; ++i) {
            const float4 w = wr[i * 256 + tid];
            acc[i].x += v * w.x;
            acc[i].y += v * w.y;
            acc[i].z += v * w.z;
            acc[i].w += v * w.w;
        }
    }

    float4* orow = (float4*)(out + row * DDIM);
    const float4* pb4 = (const float4*)pre_bias;
    #pragma unroll
    for (int i = 0; i < 4; ++i) {
        const int c4 = i * 256 + tid;
        const float4 pb = pb4[c4];
        float4 r;
        r.x = acc[i].x + pb.x;
        r.y = acc[i].y + pb.y;
        r.z = acc[i].z + pb.z;
        r.w = acc[i].w + pb.w;
        orow[c4] = r;
    }
}

// ---------------------------------------------------------------------------
// launcher
// ---------------------------------------------------------------------------
extern "C" void kernel_launch(void* const* d_in, const int* in_sizes, int n_in,
                              void* d_out, int out_size, void* d_ws, size_t ws_size,
                              hipStream_t stream) {
    (void)in_sizes; (void)n_in; (void)out_size; (void)ws_size;

    const float* x        = (const float*)d_in[0];
    // d_in[1] = ema_frequency_counter (unused in reconstruct path)
    const float* W_enc    = (const float*)d_in[2];
    const float* W_dec    = (const float*)d_in[3];
    const float* pre_bias = (const float*)d_in[4];
    const float* lat_bias = (const float*)d_in[5];
    // d_in[6] = k (fixed at 64)

    // workspace layout:
    //   acts f32 [B*D]           : 256 MB
    //   xbf  bf16 [B*D]          : 128 MB
    //   wt   bf16 [D*D] (transp) :  32 MB
    //   vals f32 [B*64]          :   4 MB
    //   idxs i32 [B*64]          :   4 MB
    char* ws = (char*)d_ws;
    float*    acts = (float*)ws;                    ws += (size_t)BDIM * DDIM * sizeof(float);
    ushort_t* xbf  = (ushort_t*)ws;                 ws += (size_t)BDIM * DDIM * sizeof(ushort_t);
    ushort_t* wt   = (ushort_t*)ws;                 ws += (size_t)DDIM * DDIM * sizeof(ushort_t);
    float*    vals = (float*)ws;                    ws += (size_t)BDIM * KTOP * sizeof(float);
    int*      idxs = (int*)ws;

    prep_x_kernel<<<(BDIM * (size_t)DDIM) / (8 * 256), 256, 0, stream>>>(x, pre_bias, xbf);
    prep_w_kernel<<<dim3(DDIM / 32, DDIM / 32), 256, 0, stream>>>(W_enc, wt);

    dim3 g1(DDIM / 128, BDIM / 128);   // (32, 128)
    enc_gemm_kernel<<<g1, 256, 0, stream>>>(xbf, wt, lat_bias, acts);

    topk_kernel<<<BDIM, 256, 0, stream>>>(acts, vals, idxs);
    dec_gemm_kernel<<<BDIM, 256, 0, stream>>>(vals, idxs, W_dec, pre_bias, (float*)d_out);
}
